// LinearAttention_57750130262732
// MI455X (gfx1250) — compile-verified, matrix operand order checked by reading, not run on hardware
//
#include <hip/hip_runtime.h>
#include <stdint.h>

typedef __attribute__((ext_vector_type(16))) __bf16 v16bf;
typedef __attribute__((ext_vector_type(8)))  float  v8f;

#define WMMA_BF16(A, B, C) \
  __builtin_amdgcn_wmma_f32_16x16x32_bf16(false, (A), false, (B), (short)0, (C), false, false)

#define BATCH    32
#define CIN      256
#define NSP      4096   // 64*64
#define QKV_O    384
#define HEADS    4
#define HEAD_DIM 32
#define COUT     256
#define CMID     128
#define SCALE    0.17677669529663687f  // 32^-0.5

// ---------------------------------------------------------------------------
// Kernel 1: qkv[b][384][4096] (bf16) = w_qkv(384x256) @ x_b(256x4096)
// macro tile 128x128, 8 waves of 64x32, K step 32 (one bf16 WMMA)
// ---------------------------------------------------------------------------
__global__ __launch_bounds__(256) void qkv_gemm(const float* __restrict__ x,
                                                const float* __restrict__ wqkv,
                                                __bf16* __restrict__ qkv) {
  const int bt   = blockIdx.z;
  const int row0 = blockIdx.x * 128;   // 0..2
  const int col0 = blockIdx.y * 128;   // 0..31
  const int tid  = threadIdx.x;
  const int lane = tid & 31;
  const int wave = tid >> 5;

  __shared__ __attribute__((aligned(32))) __bf16 Asm[128][32];  // [m][k]
  __shared__ __attribute__((aligned(32))) __bf16 Bsm[128][32];  // [n][k] transposed

  const float* xb = x + (size_t)bt * CIN * NSP;
  const int wm = (wave >> 2) * 64;   // wave row offset in macro tile
  const int wn = (wave & 3) * 32;    // wave col offset

  v8f acc[4][2];
  for (int i = 0; i < 4; ++i)
    for (int j = 0; j < 2; ++j) acc[i][j] = {};

  const int kb = (lane >> 4) * 16;
  const int ml = lane & 15;

  for (int k0 = 0; k0 < CIN; k0 += 32) {
    // issue all 8 staging loads first so they overlap, then convert to bf16
    float4 fa[4], fb[4];
#pragma unroll
    for (int it = 0; it < 4; ++it) {
      const int s = tid + it * 256;
      const int r = s >> 3, c4 = (s & 7) * 4;
      fa[it] = *(const float4*)&wqkv[(size_t)(row0 + r) * CIN + k0 + c4];
    }
#pragma unroll
    for (int it = 0; it < 4; ++it) {
      const int s = tid + it * 256;
      const int r = s >> 5, c4 = (s & 31) * 4;
      fb[it] = *(const float4*)&xb[(size_t)(k0 + r) * NSP + col0 + c4];
    }
#pragma unroll
    for (int it = 0; it < 4; ++it) {
      const int s = tid + it * 256;
      const int r = s >> 3, c4 = (s & 7) * 4;
      Asm[r][c4 + 0] = (__bf16)fa[it].x;
      Asm[r][c4 + 1] = (__bf16)fa[it].y;
      Asm[r][c4 + 2] = (__bf16)fa[it].z;
      Asm[r][c4 + 3] = (__bf16)fa[it].w;
    }
#pragma unroll
    for (int it = 0; it < 4; ++it) {
      const int s = tid + it * 256;
      const int r = s >> 5, c4 = (s & 31) * 4;
      Bsm[c4 + 0][r] = (__bf16)fb[it].x;
      Bsm[c4 + 1][r] = (__bf16)fb[it].y;
      Bsm[c4 + 2][r] = (__bf16)fb[it].z;
      Bsm[c4 + 3][r] = (__bf16)fb[it].w;
    }
    // prefetch next K tile of the streaming operand (global_prefetch_b8)
    if (k0 + 32 < CIN) {
      const int r = tid >> 3;
      __builtin_prefetch(&xb[(size_t)(k0 + 32 + r) * NSP + col0 + (tid & 7) * 16], 0, 1);
    }
    __syncthreads();

    v16bf afr[4], bfr[2];
    for (int i = 0; i < 4; ++i) afr[i] = *(const v16bf*)&Asm[wm + i * 16 + ml][kb];
    for (int j = 0; j < 2; ++j) bfr[j] = *(const v16bf*)&Bsm[wn + j * 16 + ml][kb];
    for (int i = 0; i < 4; ++i)
      for (int j = 0; j < 2; ++j)
        acc[i][j] = WMMA_BF16(afr[i], bfr[j], acc[i][j]);
    __syncthreads();
  }

  __bf16* ob = qkv + (size_t)bt * QKV_O * NSP;
  const int mh = (lane >> 4) * 8;
  for (int i = 0; i < 4; ++i)
    for (int j = 0; j < 2; ++j)
      for (int e = 0; e < 8; ++e) {
        const int m = row0 + wm + i * 16 + mh + e;
        const int n = col0 + wn + j * 16 + ml;
        ob[(size_t)m * NSP + n] = (__bf16)acc[i][j][e];
      }
}

// ---------------------------------------------------------------------------
// Kernel 2: per (batch, head): softmax(q, n), softmax(k, e),
//           context = k @ v^T (32x32), out = context^T @ q (32x4096) -> attn bf16
// ---------------------------------------------------------------------------
__global__ __launch_bounds__(256) void attn_head(__bf16* __restrict__ qkv,
                                                 __bf16* __restrict__ attn) {
  const int h = blockIdx.x;
  const int b = blockIdx.y;
  const int tid = threadIdx.x, lane = tid & 31, wave = tid >> 5;

  __bf16* qp = qkv + (size_t)b * QKV_O * NSP + (size_t)(h * HEAD_DIM) * NSP;
  __bf16* kp = qkv + (size_t)b * QKV_O * NSP + (size_t)(128 + h * HEAD_DIM) * NSP;
  __bf16* vp = qkv + (size_t)b * QKV_O * NSP + (size_t)(256 + h * HEAD_DIM) * NSP;

  __shared__ float red[256];
  __shared__ float rowstat[32];
  __shared__ __attribute__((aligned(32))) float  ctxp[2][32][32];
  __shared__ __attribute__((aligned(32))) __bf16 ctxT[32][32];  // context^T [d][e]

  // ---- step 1: softmax over n for each q row (8 threads per row) ----
  {
    const int row = tid >> 3, sub = tid & 7;
    __bf16* qr = qp + (size_t)row * NSP;
    const int n0 = sub * 512, n1 = n0 + 512;
    float mx = -3.0e38f;
    for (int n = n0; n < n1; ++n) mx = fmaxf(mx, (float)qr[n]);
    red[tid] = mx;
    __syncthreads();
    if (sub == 0) {
      float m = red[tid];
      for (int i = 1; i < 8; ++i) m = fmaxf(m, red[tid + i]);
      rowstat[row] = m;
    }
    __syncthreads();
    const float rm = rowstat[row];
    float s = 0.f;
    for (int n = n0; n < n1; ++n) s += __expf((float)qr[n] - rm);
    red[tid] = s;
    __syncthreads();
    if (sub == 0) {
      float t = red[tid];
      for (int i = 1; i < 8; ++i) t += red[tid + i];
      rowstat[row] = t;
    }
    __syncthreads();
    const float inv = SCALE / rowstat[row];
    for (int n = n0; n < n1; ++n)
      qr[n] = (__bf16)(__expf((float)qr[n] - rm) * inv);
    __threadfence();
    __syncthreads();
  }

  // ---- step 2: softmax over e (32) for each k column (16 cols per thread) ----
  for (int nn = 0; nn < 16; ++nn) {
    const int n = tid * 16 + nn;
    float mx = -3.0e38f;
    for (int e = 0; e < 32; ++e) mx = fmaxf(mx, (float)kp[(size_t)e * NSP + n]);
    float s = 0.f;
    for (int e = 0; e < 32; ++e) s += __expf((float)kp[(size_t)e * NSP + n] - mx);
    const float inv = 1.0f / s;
    for (int e = 0; e < 32; ++e) {
      const size_t idx = (size_t)e * NSP + n;
      kp[idx] = (__bf16)(__expf((float)kp[idx] - mx) * inv);
    }
  }
  __threadfence();
  __syncthreads();

  // ---- step 3: context[e][d] = sum_n k[e][n] * v[d][n] via WMMA ----
  {
    const int ti = wave & 1, tj = (wave >> 1) & 1, kh = wave >> 2;
    const int kb = (lane >> 4) * 16;
    const int ml = lane & 15;
    const __bf16* arow = kp + (size_t)(ti * 16 + ml) * NSP;  // A row = e
    const __bf16* brow = vp + (size_t)(tj * 16 + ml) * NSP;  // B col = d
    v8f acc = {};
    const int nbeg = kh * 2048, nend = nbeg + 2048;
    for (int n0 = nbeg; n0 < nend; n0 += 32) {
      const v16bf a  = *(const v16bf*)&arow[n0 + kb];
      const v16bf bm = *(const v16bf*)&brow[n0 + kb];
      acc = WMMA_BF16(a, bm, acc);
    }
    const int mh = (lane >> 4) * 8;
    for (int e = 0; e < 8; ++e)
      ctxp[kh][ti * 16 + mh + e][tj * 16 + ml] = acc[e];
  }
  __syncthreads();
  for (int s = tid; s < 1024; s += 256) {
    const int e = s >> 5, d = s & 31;
    ctxT[d][e] = (__bf16)(ctxp[0][e][d] + ctxp[1][e][d]);
  }
  __syncthreads();

  // ---- step 4: out[d][n] = sum_e ctxT[d][e] * q[e][n] (K=32 -> one WMMA) ----
  {
    __bf16* ob = attn + (size_t)b * CMID * NSP + (size_t)(h * HEAD_DIM) * NSP;
    const int kb = (lane >> 4) * 16;
    const int ml = lane & 15;
    const int mh = (lane >> 4) * 8;
    v16bf afr0 = *(const v16bf*)&ctxT[ml][kb];
    v16bf afr1 = *(const v16bf*)&ctxT[16 + ml][kb];
    const int nbeg = wave * 512, nend = nbeg + 512;
    for (int nt = nbeg; nt < nend; nt += 16) {
      const int n = nt + ml;
      v16bf bf;
      for (int j = 0; j < 16; ++j) bf[j] = qp[(size_t)(kb + j) * NSP + n];
      v8f o0 = {}, o1 = {};
      o0 = WMMA_BF16(afr0, bf, o0);
      o1 = WMMA_BF16(afr1, bf, o1);
      for (int e = 0; e < 8; ++e) {
        ob[(size_t)(mh + e) * NSP + n]      = (__bf16)o0[e];
        ob[(size_t)(16 + mh + e) * NSP + n] = (__bf16)o1[e];
      }
    }
  }
}

// ---------------------------------------------------------------------------
// Kernel 3: out[b][256][4096] (f32) = w_out(256x128) @ attn_b(128x4096) + bias
// B tile is already bf16 -> use ASYNC global->LDS copy (ASYNCcnt), then
// LDS-side transpose into the WMMA-friendly [n][k] layout.
// ---------------------------------------------------------------------------
__global__ __launch_bounds__(256) void out_gemm(const __bf16* __restrict__ attn,
                                                const float* __restrict__ wout,
                                                const float* __restrict__ bout,
                                                float* __restrict__ out) {
  const int bt   = blockIdx.z;
  const int row0 = blockIdx.x * 128;  // 0..1
  const int col0 = blockIdx.y * 128;  // 0..31
  const int tid  = threadIdx.x;
  const int lane = tid & 31;
  const int wave = tid >> 5;

  __shared__ __attribute__((aligned(32))) __bf16 Asm[128][32];
  __shared__ __attribute__((aligned(32))) __bf16 Bsm[128][32];     // [n][k]
  __shared__ __attribute__((aligned(32))) __bf16 Bstage[32][128];  // [k][n] raw

  const __bf16* ab = attn + (size_t)bt * CMID * NSP;
  const int wm = (wave >> 2) * 64;
  const int wn = (wave & 3) * 32;

  v8f acc[4][2];
  for (int i = 0; i < 4; ++i)
    for (int j = 0; j < 2; ++j) acc[i][j] = {};

  const int kb = (lane >> 4) * 16;
  const int ml = lane & 15;

  for (int k0 = 0; k0 < CMID; k0 += 32) {
    // ---- async copy B tile (32 x 128 bf16 = 8KB) straight into LDS ----
#pragma unroll
    for (int it = 0; it < 2; ++it) {
      const int s = tid + it * 256;       // 512 chunks of 16B
      const int r = s >> 4;               // k row 0..31
      const int cb = (s & 15) * 8;        // col element, step 8 bf16 = 16B
      const __bf16* src = ab + (size_t)(k0 + r) * NSP + col0 + cb;
      const uint64_t ga = (uint64_t)(uintptr_t)src;
      const uint32_t la = (uint32_t)(uintptr_t)(void*)&Bstage[r][cb];
      asm volatile("global_load_async_to_lds_b128 %0, %1, off"
                   :: "v"(la), "v"(ga) : "memory");
    }
    // ---- A tile: batched loads then native bf16 convert ----
    float4 fa[4];
#pragma unroll
    for (int it = 0; it < 4; ++it) {
      const int s = tid + it * 256;
      const int r = s >> 3, c4 = (s & 7) * 4;
      fa[it] = *(const float4*)&wout[(size_t)(row0 + r) * CMID + k0 + c4];
    }
#pragma unroll
    for (int it = 0; it < 4; ++it) {
      const int s = tid + it * 256;
      const int r = s >> 3, c4 = (s & 7) * 4;
      Asm[r][c4 + 0] = (__bf16)fa[it].x;
      Asm[r][c4 + 1] = (__bf16)fa[it].y;
      Asm[r][c4 + 2] = (__bf16)fa[it].z;
      Asm[r][c4 + 3] = (__bf16)fa[it].w;
    }
    if (k0 + 32 < CMID) {
      const int r = tid >> 3;
      __builtin_prefetch(&ab[(size_t)(k0 + 32 + r) * NSP + col0 + (tid & 7) * 16], 0, 1);
    }
    asm volatile("s_wait_asynccnt 0x0" ::: "memory");
    __syncthreads();

    // ---- LDS-side transpose Bstage[k][n] -> Bsm[n][k] ----
#pragma unroll
    for (int it = 0; it < 2; ++it) {
      const int s = tid + it * 256;
      const int r = s >> 4, cb = (s & 15) * 8;
#pragma unroll
      for (int j = 0; j < 8; ++j) Bsm[cb + j][r] = Bstage[r][cb + j];
    }
    __syncthreads();

    v16bf afr[4], bfr[2];
    for (int i = 0; i < 4; ++i) afr[i] = *(const v16bf*)&Asm[wm + i * 16 + ml][kb];
    for (int j = 0; j < 2; ++j) bfr[j] = *(const v16bf*)&Bsm[wn + j * 16 + ml][kb];
    for (int i = 0; i < 4; ++i)
      for (int j = 0; j < 2; ++j)
        acc[i][j] = WMMA_BF16(afr[i], bfr[j], acc[i][j]);
    __syncthreads();
  }

  float* ob = out + (size_t)bt * COUT * NSP;
  const int mh = (lane >> 4) * 8;
  for (int i = 0; i < 4; ++i)
    for (int j = 0; j < 2; ++j)
      for (int e = 0; e < 8; ++e) {
        const int m = row0 + wm + i * 16 + mh + e;
        const int n = col0 + wn + j * 16 + ml;
        ob[(size_t)m * NSP + n] = acc[i][j][e] + bout[m];
      }
}

// ---------------------------------------------------------------------------
extern "C" void kernel_launch(void* const* d_in, const int* in_sizes, int n_in,
                              void* d_out, int out_size, void* d_ws, size_t ws_size,
                              hipStream_t stream) {
  const float* x    = (const float*)d_in[0];
  const float* wqkv = (const float*)d_in[1];
  const float* wout = (const float*)d_in[2];
  const float* bout = (const float*)d_in[3];
  float* out = (float*)d_out;

  // ws layout: qkv bf16 (32*384*4096*2 = 100663296 B) | attn bf16 (32*128*4096*2 = 33554432 B)
  __bf16* qkv  = (__bf16*)d_ws;
  __bf16* attn = (__bf16*)((char*)d_ws + (size_t)BATCH * QKV_O * NSP * 2);

  dim3 g1(QKV_O / 128, NSP / 128, BATCH);
  qkv_gemm<<<g1, 256, 0, stream>>>(x, wqkv, qkv);

  dim3 g2(HEADS, BATCH);
  attn_head<<<g2, 256, 0, stream>>>(qkv, attn);

  dim3 g3(COUT / 128, NSP / 128, BATCH);
  out_gemm<<<g3, 256, 0, stream>>>(attn, wout, bout, out);
}
